// MoE_18159121727733
// MI455X (gfx1250) — matrix operand order repaired and checked by reading, not compiled, not run on hardware
//
#include <hip/hip_runtime.h>

// MoE top-2 routed implementation for gfx1250 (CDNA5, wave32, WMMA bf16).
// E=8 D=1024 O=1024 G=512 H=2048 K=2 B=8192
//
// Pipeline:
//  1) transpose_bf16_kernel: W1,W2,Wg1 fp32 -> transposed bf16 so WMMA
//     B-fragments are single contiguous 32B global loads.
//  2) gate_kernel: WMMA bf16 gate layer1 + fp32 VALU layer2, softmax, top-2,
//     entropy partials, per-expert bucket push.
//  3) expert_kernel: per (expert, 16-token tile) 2-layer MLP, all WMMA bf16.
//     k-outer / n-tile-inner loops: A fragment loaded once per k-step, 4-8
//     independent WMMA accumulation chains for latency hiding.
//  4) combine_kernel + aux_kernel.

#define E_ 8
#define D_ 1024
#define O_ 1024
#define G_ 512
#define H_ 2048
#define B_ 8192
#define NGATE (B_ / 16)       // 512 gate blocks
#define TILES_PER_E (B_ / 16) // 512 tiles per expert (worst case)

typedef __bf16 bf16;
typedef __attribute__((ext_vector_type(16))) __bf16 v16bf;
typedef __attribute__((ext_vector_type(8)))  __bf16 v8bf;
typedef __attribute__((ext_vector_type(8)))  float  v8f;

// ---- WMMA fragment helpers (k order used: lane half h holds k = 16h..16h+15 for both A and B) ----

// A (16x32 bf16) from LDS row-major tile. Lane L<16: row L, K = k0+{0..15};
// lane L>=16: row L-16, K = k0+{16..31}.  rowk0 = &tile[row*ld + k0].
// Lowers to 2x ds_load_b128.
__device__ inline v16bf a_frag_lds(const bf16* rowk0, int lane) {
  const int kb = (lane & 16) ? 16 : 0;
  v8bf lo = *(const v8bf*)(rowk0 + kb);
  v8bf hi = *(const v8bf*)(rowk0 + kb + 8);
  v16bf a;
#pragma unroll
  for (int i = 0; i < 8; ++i) { a[i] = lo[i]; a[i + 8] = hi[i]; }
  return a;
}

// B (32x16 bf16) from TRANSPOSED bf16 weights Wt[n][k]: 16 contiguous K per lane
// (lane<16: col n0+lane, K k0..k0+15; lane>=16: col n0+lane-16, K k0+16..31).
// One 32B contiguous load -> 2x global_load_b128.
__device__ inline v16bf b_frag_t(const bf16* p) {
  return *(const v16bf*)p;
}

__device__ inline v8f wmma_bf16(v16bf a, v16bf b, v8f c) {
  return __builtin_amdgcn_wmma_f32_16x16x32_bf16(false, a, false, b, (short)0, c,
                                                 false, false);
}

// ---------------------------------------------------------------------------
__global__ void init_kernel(int* cnt) {
  if (blockIdx.x == 0 && threadIdx.x < E_) cnt[threadIdx.x] = 0;
}

// ---------------------------------------------------------------------------
// Tiled transpose + fp32->bf16: src [batch][K][N] fp32 -> dst [batch][N][K] bf16.
__global__ __launch_bounds__(256)
void transpose_bf16_kernel(const float* __restrict__ src, bf16* __restrict__ dst,
                           int K, int N) {
  __shared__ bf16 t[32][33];
  const int nt = blockIdx.x, kt = blockIdx.y, b = blockIdx.z;
  const float* s = src + (size_t)b * K * N;
  bf16* d = dst + (size_t)b * K * N;
  const int tid = threadIdx.x;
  const int c = tid & 31, r0 = tid >> 5;
#pragma unroll
  for (int i = 0; i < 4; ++i) {
    int r = r0 + i * 8;
    t[r][c] = (bf16)s[(size_t)(kt * 32 + r) * N + (nt * 32 + c)];
  }
  __syncthreads();
#pragma unroll
  for (int i = 0; i < 4; ++i) {
    int r = r0 + i * 8;  // row within the N-dim of dst
    d[(size_t)(nt * 32 + r) * K + (kt * 32 + c)] = t[c][r];
  }
}

// ---------------------------------------------------------------------------
// Gate: logits = relu(x@Wg1+bg1)@Wg2+bg2; softmax; top-2; entropy; buckets.
__global__ __launch_bounds__(256)
void gate_kernel(const float* __restrict__ x,
                 const bf16* __restrict__ Wg1t, const float* __restrict__ bg1,
                 const float* __restrict__ Wg2, const float* __restrict__ bg2,
                 float* __restrict__ topW, int* __restrict__ bucket,
                 int* __restrict__ cnt, float* __restrict__ epart) {
  __shared__ __align__(16) bf16 sx[16 * D_];   // 32 KB
  __shared__ __align__(16) bf16 shg[16 * G_];  // 16 KB
  __shared__ float slog[16 * E_];
  __shared__ float sent[16];

  const int tid  = threadIdx.x;
  const int lane = tid & 31;
  const int w    = tid >> 5;
  const int t0   = blockIdx.x * 16;

#pragma unroll 4
  for (int it = 0; it < (16 * D_) / 256; ++it) {
    int idx = it * 256 + tid;
    int m = idx >> 10, d = idx & (D_ - 1);
    sx[idx] = (bf16)x[(size_t)(t0 + m) * D_ + d];
  }
  __syncthreads();

  const int mrow   = lane & 15;
  const int rowoff = (lane & 16) ? 8 : 0;
  const int koff   = (lane & 16) ? 16 : 0;
  const int ncol   = lane & 15;

  // GEMM1: [16,1024]x[1024,512] -> relu -> shg. 4 N-tiles per wave, k-outer so
  // one A fragment feeds 4 independent WMMA chains.
  {
    const bf16* bp0 = Wg1t + (size_t)(w * 64 + ncol) * D_ + koff;
    v8f accg[4];
#pragma unroll
    for (int j = 0; j < 4; ++j) { v8f z = {}; accg[j] = z; }
    for (int k0 = 0; k0 < D_; k0 += 32) {
      v16bf a = a_frag_lds(sx + mrow * D_ + k0, lane);
#pragma unroll
      for (int j = 0; j < 4; ++j)
        accg[j] = wmma_bf16(a, b_frag_t(bp0 + (size_t)j * 16 * D_ + k0), accg[j]);
    }
#pragma unroll
    for (int j = 0; j < 4; ++j) {
      int n = (w * 4 + j) * 16 + ncol;
      float bias = bg1[n];
#pragma unroll
      for (int r = 0; r < 8; ++r) {
        float v = accg[j][r] + bias;
        shg[(r + rowoff) * G_ + n] = (bf16)(v > 0.f ? v : 0.f);
      }
    }
  }
  __syncthreads();

  // layer 2 (G=512 -> E=8) in fp32 VALU
  if (tid < 16 * E_) {
    int m = tid >> 3, e = tid & 7;
    float s = 0.f;
    for (int g = 0; g < G_; ++g)
      s += (float)shg[m * G_ + g] * Wg2[(size_t)g * E_ + e];
    slog[m * E_ + e] = s + bg2[e];
  }
  __syncthreads();

  // softmax / entropy / top-2 / bucket push (one token per thread)
  if (tid < 16) {
    int m = tid, token = t0 + m;
    float l[E_], p[E_];
    float mx = -3.0e38f;
    for (int e = 0; e < E_; ++e) { l[e] = slog[m * E_ + e]; mx = fmaxf(mx, l[e]); }
    float sum = 0.f;
    for (int e = 0; e < E_; ++e) { l[e] = __expf(l[e] - mx); sum += l[e]; }
    float ent = 0.f;
    for (int e = 0; e < E_; ++e) {
      p[e] = l[e] / sum;
      ent -= p[e] * __logf(p[e] + 1e-9f);
    }
    int i1 = 0;
    for (int e = 1; e < E_; ++e) if (p[e] > p[i1]) i1 = e;       // ties -> lowest idx
    int i2 = (i1 == 0) ? 1 : 0;
    for (int e = 0; e < E_; ++e) if (e != i1 && p[e] > p[i2]) i2 = e;
    float denom = p[i1] + p[i2] + 1e-9f;
    topW[token * 2 + 0] = p[i1] / denom;
    topW[token * 2 + 1] = p[i2] / denom;
    int pos1 = atomicAdd(&cnt[i1], 1);
    bucket[i1 * B_ + pos1] = token;              // k = 0
    int pos2 = atomicAdd(&cnt[i2], 1);
    bucket[i2 * B_ + pos2] = token | (1 << 16);  // k = 1
    sent[m] = ent;
  }
  __syncthreads();
  if (tid == 0) {
    float s = 0.f;
    for (int m = 0; m < 16; ++m) s += sent[m];
    epart[blockIdx.x] = s;
  }
}

// ---------------------------------------------------------------------------
// Expert MLP for one (expert, 16-token tile): out = (relu(x@W1+b1)@W2+b2)*gateW
// W1t: bf16 [E][H][D]; W2t: bf16 [E][O][H]. H processed in 4 phases of 512;
// GEMM2 accumulates in 64 VGPRs per lane across phases.
__global__ __launch_bounds__(256)
void expert_kernel(const float* __restrict__ x,
                   const bf16* __restrict__ W1t, const float* __restrict__ b1,
                   const bf16* __restrict__ W2t, const float* __restrict__ b2,
                   const float* __restrict__ topW, const int* __restrict__ bucket,
                   const int* __restrict__ cnt, float* __restrict__ slotbuf) {
  const int e    = blockIdx.x >> 9;
  const int tile = blockIdx.x & 511;
  const int nTok = cnt[e];
  const int base = tile * 16;
  if (base >= nTok) return;
  const int nv = min(16, nTok - base);

  __shared__ __align__(16) bf16 sx[16 * D_];  // 32 KB (x tokens)
  __shared__ __align__(16) bf16 sh[16 * 512]; // 16 KB (hidden chunk)
  __shared__ int   sslot[16];
  __shared__ float swt[16];

  const int tid = threadIdx.x, lane = tid & 31, w = tid >> 5;

  if (tid < 16) {
    if (tid < nv) {
      int entry = bucket[e * B_ + base + tid];
      int tok = entry & 0xFFFF;
      int k   = entry >> 16;
      sslot[tid] = tok * 2 + k;               // (token, k) output slot
      swt[tid]   = topW[tok * 2 + k];
    } else { sslot[tid] = -1; swt[tid] = 0.f; }
  }
  __syncthreads();

  // gather + convert x rows
#pragma unroll 4
  for (int it = 0; it < (16 * D_) / 256; ++it) {
    int idx = it * 256 + tid;
    int m = idx >> 10, d = idx & (D_ - 1);
    int s = sslot[m];
    sx[idx] = (s >= 0) ? (bf16)x[(size_t)(s >> 1) * D_ + d] : (bf16)0.f;
  }

  const int mrow   = lane & 15;
  const int rowoff = (lane & 16) ? 8 : 0;
  const int koff   = (lane & 16) ? 16 : 0;
  const int ncol   = lane & 15;

  v8f acc2[8];
#pragma unroll
  for (int j = 0; j < 8; ++j) { v8f z = {}; acc2[j] = z; }

  for (int phase = 0; phase < 4; ++phase) {
    const int hbase = phase * 512;
    __syncthreads();  // sh free to overwrite (and sx staged on phase 0)

    // GEMM1 chunk: h[:, hbase:hbase+512] = relu(x @ W1 + b1).
    // k-outer, 4 N-tiles per wave -> one A frag, 4 independent WMMA chains.
    {
      const bf16* bp0 =
          W1t + ((size_t)e * H_ + hbase + w * 64 + ncol) * D_ + koff;
      v8f accg[4];
#pragma unroll
      for (int j = 0; j < 4; ++j) { v8f z = {}; accg[j] = z; }
      for (int k0 = 0; k0 < D_; k0 += 32) {
        v16bf a = a_frag_lds(sx + mrow * D_ + k0, lane);
#pragma unroll
        for (int j = 0; j < 4; ++j)
          accg[j] = wmma_bf16(a, b_frag_t(bp0 + (size_t)j * 16 * D_ + k0), accg[j]);
      }
#pragma unroll
      for (int j = 0; j < 4; ++j) {
        int n0   = (w * 4 + j) * 16;
        int hcol = hbase + n0 + ncol;
        float bias = b1[(size_t)e * H_ + hcol];
#pragma unroll
        for (int r = 0; r < 8; ++r) {
          float v = accg[j][r] + bias;
          sh[(r + rowoff) * 512 + n0 + ncol] = (bf16)(v > 0.f ? v : 0.f);
        }
      }
    }
    __syncthreads();

    // GEMM2 partial: out += h_chunk @ W2[hbase:hbase+512, :].
    // k-outer, 8 N-tiles per wave -> one A frag, 8 independent WMMA chains.
    {
      const bf16* bp2 =
          W2t + ((size_t)e * O_ + w * 128 + ncol) * H_ + hbase + koff;
      for (int k0 = 0; k0 < 512; k0 += 32) {
        v16bf a = a_frag_lds(sh + mrow * 512 + k0, lane);
#pragma unroll
        for (int j = 0; j < 8; ++j)
          acc2[j] = wmma_bf16(a, b_frag_t(bp2 + (size_t)j * 16 * H_ + k0), acc2[j]);
      }
    }
  }

  // bias, gate-scale, write per-(token,k) slot
#pragma unroll
  for (int j = 0; j < 8; ++j) {
    int ocol = (w * 8 + j) * 16 + ncol;
    float bias = b2[(size_t)e * O_ + ocol];
#pragma unroll
    for (int r = 0; r < 8; ++r) {
      int m = r + rowoff;
      int s = sslot[m];
      if (s >= 0)
        slotbuf[(size_t)s * O_ + ocol] = (acc2[j][r] + bias) * swt[m];
    }
  }
}

// ---------------------------------------------------------------------------
__global__ __launch_bounds__(256)
void combine_kernel(const float* __restrict__ slotbuf, float* __restrict__ out) {
  size_t i = (size_t)blockIdx.x * 256 + threadIdx.x;
  size_t b = i >> 10, o = i & (O_ - 1);
  out[i] = slotbuf[(b * 2) * O_ + o] + slotbuf[(b * 2 + 1) * O_ + o];
}

__global__ void aux_kernel(const int* __restrict__ cnt,
                           const float* __restrict__ epart,
                           float* __restrict__ out) {
  if (blockIdx.x != 0 || threadIdx.x != 0) return;
  float es = 0.f;
  for (int i = 0; i < NGATE; ++i) es += epart[i];       // fixed order: deterministic
  float load[E_], mean = 0.f;
  for (int e = 0; e < E_; ++e) {
    load[e] = (float)cnt[e] / ((float)B_ + 1e-9f);
    mean += load[e];
  }
  mean /= (float)E_;
  float var = 0.f;
  for (int e = 0; e < E_; ++e) { float d = load[e] - mean; var += d * d; }
  var /= (float)(E_ - 1);                                // unbiased (torch var)
  out[(size_t)B_ * O_] = var + es / (float)B_;
}

// ---------------------------------------------------------------------------
extern "C" void kernel_launch(void* const* d_in, const int* in_sizes, int n_in,
                              void* d_out, int out_size, void* d_ws, size_t ws_size,
                              hipStream_t stream) {
  const float* x   = (const float*)d_in[0];
  const float* W1  = (const float*)d_in[1];
  const float* b1  = (const float*)d_in[2];
  const float* W2  = (const float*)d_in[3];
  const float* b2  = (const float*)d_in[4];
  const float* Wg1 = (const float*)d_in[5];
  const float* bg1 = (const float*)d_in[6];
  const float* Wg2 = (const float*)d_in[7];
  const float* bg2 = (const float*)d_in[8];
  float* out = (float*)d_out;

  // workspace layout (~136 MB): slot buffer + transposed bf16 weights + gate meta
  char* ws = (char*)d_ws;
  size_t off = 0;
  float* slotbuf = (float*)(ws + off); off += (size_t)B_ * 2 * O_ * sizeof(float);
  bf16* W1t  = (bf16*)(ws + off);      off += (size_t)E_ * D_ * H_ * sizeof(bf16);
  bf16* W2t  = (bf16*)(ws + off);      off += (size_t)E_ * H_ * O_ * sizeof(bf16);
  bf16* Wg1t = (bf16*)(ws + off);      off += (size_t)D_ * G_ * sizeof(bf16);
  float* topW = (float*)(ws + off);    off += (size_t)B_ * 2 * sizeof(float);
  int* bucket = (int*)(ws + off);      off += (size_t)E_ * B_ * sizeof(int);
  int* cnt    = (int*)(ws + off);      off += 64 * sizeof(int);
  float* epart = (float*)(ws + off);

  init_kernel<<<1, 32, 0, stream>>>(cnt);

  // weight pre-pass: fp32 [b][K][N] -> bf16 [b][N][K]
  transpose_bf16_kernel<<<dim3(H_ / 32, D_ / 32, E_), 256, 0, stream>>>(W1, W1t, D_, H_);
  transpose_bf16_kernel<<<dim3(O_ / 32, H_ / 32, E_), 256, 0, stream>>>(W2, W2t, H_, O_);
  transpose_bf16_kernel<<<dim3(G_ / 32, D_ / 32, 1), 256, 0, stream>>>(Wg1, Wg1t, D_, G_);

  gate_kernel<<<NGATE, 256, 0, stream>>>(x, Wg1t, bg1, Wg2, bg2,
                                         topW, bucket, cnt, epart);
  expert_kernel<<<E_ * TILES_PER_E, 256, 0, stream>>>(x, W1t, b1, W2t, b2,
                                                      topW, bucket, cnt, slotbuf);
  combine_kernel<<<(B_ * O_) / 256, 256, 0, stream>>>(slotbuf, out);
  aux_kernel<<<1, 32, 0, stream>>>(cnt, epart, out);
}
